// FrustumOffsetNetwork_30700426232355
// MI455X (gfx1250) — matrix operand order repaired and checked by reading, not compiled, not run on hardware
//
#include <hip/hip_runtime.h>
#include <hip/hip_bf16.h>

#define HH 128
#define WWID 352
#define CC 128
#define BB 2

typedef float v8f __attribute__((ext_vector_type(8)));
typedef __bf16 v16bf __attribute__((ext_vector_type(16)));

union Frag16 {
    v16bf v;
    unsigned int u32[8];
    unsigned short u16[16];
};

union PackBF8 {
    unsigned short s[8];
    uint4 q;
};

static __device__ __forceinline__ unsigned short f2bf(float f) {
    unsigned int u = __float_as_uint(f);
    unsigned int r = u + 0x7FFFu + ((u >> 16) & 1u);
    return (unsigned short)(r >> 16);
}
static __device__ __forceinline__ float bf2f(unsigned short h) {
    return __uint_as_float(((unsigned int)h) << 16);
}

static __device__ __forceinline__ void async_copy_b128(unsigned ldsOff,
                                                       unsigned long long gaddr) {
    asm volatile("global_load_async_to_lds_b128 %0, %1, off"
                 :
                 : "v"(ldsOff), "v"(gaddr)
                 : "memory");
}
static __device__ __forceinline__ void wait_asynccnt0() {
    asm volatile("s_wait_asynccnt 0" ::: "memory");
}

// ---------------------------------------------------------------------------
// Prep: NHWC bf16 input, weight repack [tap][co][ci], BN fold, loss zero.
// ---------------------------------------------------------------------------
__global__ __launch_bounds__(256) void prep_kernel(
    const float* __restrict__ x, const float* __restrict__ w1, const float* __restrict__ w2,
    const float* __restrict__ g1, const float* __restrict__ b1v, const float* __restrict__ m1,
    const float* __restrict__ v1,
    const float* __restrict__ g2, const float* __restrict__ b2v, const float* __restrict__ m2,
    const float* __restrict__ v2,
    const float* __restrict__ wf,
    unsigned short* __restrict__ xin, unsigned short* __restrict__ wt1,
    unsigned short* __restrict__ wt2, unsigned short* __restrict__ wfb,
    float* __restrict__ sc1, float* __restrict__ sh1,
    float* __restrict__ sc2, float* __restrict__ sh2,
    float* __restrict__ lossPtr, unsigned long long nxc)
{
    unsigned long long i = (unsigned long long)blockIdx.x * 256ull + threadIdx.x;
    const unsigned long long WTN = 9ull * CC * CC;
    if (i < nxc) {
        // NHWC destination: i = ((b*HH+y)*WWID+x)*CC + c   from NCHW source
        unsigned long long c = i & 127;
        unsigned long long p = i >> 7;
        unsigned long long xg = p % WWID;
        unsigned long long t = p / WWID;
        unsigned long long yg = t % HH;
        unsigned long long bg = t / HH;
        xin[i] = f2bf(x[((bg * CC + c) * HH + yg) * WWID + xg]);
        return;
    }
    i -= nxc;
    if (i < WTN) {
        unsigned long long tap = i / (CC * CC), r = i % (CC * CC);
        unsigned long long co = r >> 7, ci = r & 127;
        wt1[i] = f2bf(w1[(co * CC + ci) * 9ull + tap]);
        return;
    }
    i -= WTN;
    if (i < WTN) {
        unsigned long long tap = i / (CC * CC), r = i % (CC * CC);
        unsigned long long co = r >> 7, ci = r & 127;
        wt2[i] = f2bf(w2[(co * CC + ci) * 9ull + tap]);
        return;
    }
    i -= WTN;
    if (i < (unsigned long long)CC * CC) { wfb[i] = f2bf(wf[i]); return; }
    i -= (unsigned long long)CC * CC;
    if (i < CC) {
        float inv = g1[i] / sqrtf(v1[i] + 1e-5f);
        sc1[i] = inv; sh1[i] = b1v[i] - m1[i] * inv;
        return;
    }
    i -= CC;
    if (i < CC) {
        float inv = g2[i] / sqrtf(v2[i] + 1e-5f);
        sc2[i] = inv; sh2[i] = b2v[i] - m2[i] * inv;
        return;
    }
    i -= CC;
    if (i == 0) *lossPtr = 0.f;
}

// ---------------------------------------------------------------------------
// 3x3 conv (zero pad) + BN + ReLU as implicit GEMM with WMMA bf16.
// Activations NHWC bf16. Patch staged with async global->LDS DMA.
// Block: 256 thr (8 waves). Tile: 128 co x 32 px. wt layout: [tap][co][ci].
// ---------------------------------------------------------------------------
__global__ __launch_bounds__(256) void conv3_bn_relu_kernel(
    const unsigned short* __restrict__ xin, const unsigned short* __restrict__ wt,
    const float* __restrict__ sc, const float* __restrict__ sh,
    unsigned short* __restrict__ out)
{
    __shared__ __align__(32) unsigned short patch[3][34][CC];  // [dy][x][ci]
    const int tid = threadIdx.x;
    const int lane = tid & 31;
    const int wave = tid >> 5;
    const int x0 = blockIdx.x * 32;
    const int y = blockIdx.y;
    const int b = blockIdx.z;

    // Stage 3 slabs of 34 rows x 256B each. Rows are contiguous in BOTH
    // global (NHWC) and LDS -> pure block copy via async b128 DMA.
    // OOB chunks (zero pad) are written with zeros instead (disjoint).
    {
        const unsigned ldsBase = (unsigned)(unsigned long long)(&patch[0][0][0]);
        for (int k = tid; k < 3 * 34 * 16; k += 256) {   // 16-byte chunks
            int slab = k / (34 * 16);
            int rem = k - slab * (34 * 16);
            int xx = rem >> 4;                 // 0..33
            int sub = (rem & 15) << 4;         // byte offset within row
            int yy = y + slab - 1;
            int xg = x0 + xx - 1;
            unsigned ldsOff = ldsBase + (unsigned)(slab * 34 * 256 + (xx << 8) + sub);
            if (yy >= 0 && yy < HH && xg >= 0 && xg < WWID) {
                const unsigned short* gp =
                    xin + (((size_t)b * HH + yy) * WWID + xg) * CC;
                async_copy_b128(ldsOff, (unsigned long long)gp + (unsigned)sub);
            } else {
                uint4* zp = (uint4*)((char*)&patch[0][0][0] +
                                     (size_t)(slab * 34 * 256 + (xx << 8) + sub));
                *zp = uint4{0u, 0u, 0u, 0u};
            }
        }
    }
    wait_asynccnt0();
    __syncthreads();

    const int coBase = wave * 16;
    const int m = lane & 15;
    const int hi = lane >> 4;
    v8f acc0 = {0.f, 0.f, 0.f, 0.f, 0.f, 0.f, 0.f, 0.f};
    v8f acc1 = {0.f, 0.f, 0.f, 0.f, 0.f, 0.f, 0.f, 0.f};

    for (int cb = 0; cb < 4; ++cb) {
        const int cbase = cb * 32 + hi * 16;
        const int h2 = hi * 4;
        #pragma unroll
        for (int tap = 0; tap < 9; ++tap) {
            const int dy = tap / 3, dx = tap % 3;
            Frag16 a, bf0, bf1;
            const unsigned int* pa =
                (const unsigned int*)(wt + ((size_t)tap * CC + coBase + m) * CC + cb * 32);
            #pragma unroll
            for (int v = 0; v < 4; ++v) { a.u32[v] = pa[2 * h2 + v]; a.u32[4 + v] = pa[2 * h2 + 4 + v]; }
            const unsigned int* pb0 = (const unsigned int*)(&patch[dy][m + dx][cbase]);
            const unsigned int* pb1 = (const unsigned int*)(&patch[dy][m + 16 + dx][cbase]);
            #pragma unroll
            for (int v = 0; v < 8; ++v) { bf0.u32[v] = pb0[v]; bf1.u32[v] = pb1[v]; }
            acc0 = __builtin_amdgcn_wmma_f32_16x16x32_bf16(false, a.v, false, bf0.v,
                                                           (short)0, acc0, false, false);
            acc1 = __builtin_amdgcn_wmma_f32_16x16x32_bf16(false, a.v, false, bf1.v,
                                                           (short)0, acc1, false, false);
        }
    }

    // BN + ReLU + NHWC b128 stores (8 contiguous co per lane).
    PackBF8 o0, o1;
    #pragma unroll
    for (int v = 0; v < 8; ++v) {
        int co = coBase + hi * 8 + v;
        float s = sc[co], t = sh[co];
        o0.s[v] = f2bf(fmaxf(acc0[v] * s + t, 0.f));
        o1.s[v] = f2bf(fmaxf(acc1[v] * s + t, 0.f));
    }
    size_t p0 = (((size_t)b * HH + y) * WWID + x0 + m) * CC + coBase + hi * 8;
    size_t p1 = p0 + (size_t)16 * CC;
    *(uint4*)(out + p0) = o0.q;
    *(uint4*)(out + p1) = o1.q;
}

// ---------------------------------------------------------------------------
// Head: af = wf*x + bf (WMMA), hl/wl 7-ch dots, softmax -> wH/wW aggregates,
// loss atomicAdd. x2 NHWC bf16 in, af NHWC fp32 out.
// ---------------------------------------------------------------------------
__global__ __launch_bounds__(256) void head_kernel(
    const unsigned short* __restrict__ x2, const unsigned short* __restrict__ wfb,
    const float* __restrict__ bfv,
    const float* __restrict__ wh, const float* __restrict__ bh,
    const float* __restrict__ ww, const float* __restrict__ bw,
    float* __restrict__ af, float* __restrict__ wHo, float* __restrict__ wWo,
    float* __restrict__ lossPtr)
{
    __shared__ __align__(32) unsigned short xs[32][CC];  // [px][ci]
    __shared__ float hlb[7][32];
    __shared__ float wlb[7][32];
    const int tid = threadIdx.x;
    const int lane = tid & 31;
    const int wave = tid >> 5;
    const int x0 = blockIdx.x * 32;
    const int y = blockIdx.y;
    const int b = blockIdx.z;

    // 8KB tile contiguous in both global (NHWC) and LDS -> async DMA copy.
    {
        const unsigned ldsBase = (unsigned)(unsigned long long)(&xs[0][0]);
        const unsigned short* gp = x2 + (((size_t)b * HH + y) * WWID + x0) * CC;
        for (int k = tid; k < 512; k += 256) {
            async_copy_b128(ldsBase + (unsigned)(k << 4),
                            (unsigned long long)gp + (unsigned)(k << 4));
        }
    }
    wait_asynccnt0();
    __syncthreads();

    const int coBase = wave * 16;
    const int m = lane & 15;
    const int hi = lane >> 4;
    v8f acc0 = {0.f, 0.f, 0.f, 0.f, 0.f, 0.f, 0.f, 0.f};
    v8f acc1 = {0.f, 0.f, 0.f, 0.f, 0.f, 0.f, 0.f, 0.f};

    #pragma unroll
    for (int cb = 0; cb < 4; ++cb) {
        const int cbase = cb * 32 + hi * 16;
        const int h2 = hi * 4;
        Frag16 a, bf0, bf1;
        const unsigned int* pa = (const unsigned int*)(wfb + ((size_t)(coBase + m)) * CC + cb * 32);
        #pragma unroll
        for (int v = 0; v < 4; ++v) { a.u32[v] = pa[2 * h2 + v]; a.u32[4 + v] = pa[2 * h2 + 4 + v]; }
        const unsigned int* pb0 = (const unsigned int*)(&xs[m][cbase]);
        const unsigned int* pb1 = (const unsigned int*)(&xs[m + 16][cbase]);
        #pragma unroll
        for (int v = 0; v < 8; ++v) { bf0.u32[v] = pb0[v]; bf1.u32[v] = pb1[v]; }
        acc0 = __builtin_amdgcn_wmma_f32_16x16x32_bf16(false, a.v, false, bf0.v,
                                                       (short)0, acc0, false, false);
        acc1 = __builtin_amdgcn_wmma_f32_16x16x32_bf16(false, a.v, false, bf1.v,
                                                       (short)0, acc1, false, false);
    }

    // af NHWC fp32, vectorized (8 contiguous co per lane).
    {
        size_t p0 = ((size_t)(b * HH + y) * WWID + x0 + m) * CC + coBase + hi * 8;
        size_t p1 = p0 + (size_t)16 * CC;
        float4 a0, a1, b0q, b1q;
        a0.x = acc0[0] + bfv[coBase + hi * 8 + 0];
        a0.y = acc0[1] + bfv[coBase + hi * 8 + 1];
        a0.z = acc0[2] + bfv[coBase + hi * 8 + 2];
        a0.w = acc0[3] + bfv[coBase + hi * 8 + 3];
        b0q.x = acc0[4] + bfv[coBase + hi * 8 + 4];
        b0q.y = acc0[5] + bfv[coBase + hi * 8 + 5];
        b0q.z = acc0[6] + bfv[coBase + hi * 8 + 6];
        b0q.w = acc0[7] + bfv[coBase + hi * 8 + 7];
        a1.x = acc1[0] + bfv[coBase + hi * 8 + 0];
        a1.y = acc1[1] + bfv[coBase + hi * 8 + 1];
        a1.z = acc1[2] + bfv[coBase + hi * 8 + 2];
        a1.w = acc1[3] + bfv[coBase + hi * 8 + 3];
        b1q.x = acc1[4] + bfv[coBase + hi * 8 + 4];
        b1q.y = acc1[5] + bfv[coBase + hi * 8 + 5];
        b1q.z = acc1[6] + bfv[coBase + hi * 8 + 6];
        b1q.w = acc1[7] + bfv[coBase + hi * 8 + 7];
        *(float4*)(af + p0) = a0;
        *(float4*)(af + p0 + 4) = b0q;
        *(float4*)(af + p1) = a1;
        *(float4*)(af + p1 + 4) = b1q;
    }

    // hl / wl 7-channel projections
    {
        int o = tid >> 5, px = tid & 31;
        if (o < 7) {
            float hv = bh[o], wv = bw[o];
            for (int ci = 0; ci < CC; ++ci) {
                float xv = bf2f(xs[px][ci]);
                hv += xv * wh[o * CC + ci];
                wv += xv * ww[o * CC + ci];
            }
            hlb[o][px] = hv;
            wlb[o][px] = wv;
        }
    }
    __syncthreads();

    if (tid < 32) {
        int px = tid;
        float h[7], w[7];
        float mh = -1e30f, mw = -1e30f;
        for (int o = 0; o < 7; ++o) {
            h[o] = hlb[o][px]; w[o] = wlb[o][px];
            mh = fmaxf(mh, h[o]); mw = fmaxf(mw, w[o]);
        }
        float sh_ = 0.f, sw_ = 0.f;
        for (int o = 0; o < 7; ++o) {
            h[o] = __expf(h[o] - mh); sh_ += h[o];
            w[o] = __expf(w[o] - mw); sw_ += w[o];
        }
        float rh = 1.f / sh_, rw = 1.f / sw_;
        for (int o = 0; o < 7; ++o) { h[o] *= rh; w[o] *= rw; }

        float aggH[4] = {h[3] + h[4] + h[5] + h[6], h[2], h[1], h[0]};
        float aggW[4] = {w[3] + w[4] + w[5] + w[6], w[2], w[1], w[0]};
        for (int s = 0; s < 4; ++s) {
            wHo[(((size_t)b * 4 + s) * HH + y) * WWID + x0 + px] = aggH[s];
            wWo[(((size_t)b * 4 + s) * HH + y) * WWID + x0 + px] = aggW[s];
        }
        const float vals[7] = {3.f, 2.f, 1.f, 0.f, 1.f, 2.f, 3.f};
        float eh = 0.f, ew = 0.f;
        for (int o = 0; o < 7; ++o) { eh += h[o] * vals[o]; ew += w[o] * vals[o]; }
        atomicAdd(lossPtr, (eh + ew) * (0.5f / (float)(BB * HH * WWID)));
    }
}

// ---------------------------------------------------------------------------
// Soft offset: 16 clamped taps, NHWC layouts.
// ---------------------------------------------------------------------------
__global__ __launch_bounds__(256) void soft_offset_kernel(
    const float* __restrict__ af, const float* __restrict__ wH,
    const float* __restrict__ wW, float* __restrict__ aligned)
{
    size_t idx = (size_t)blockIdx.x * 256ull + threadIdx.x;
    const size_t NTOT = (size_t)BB * HH * WWID * CC;
    if (idx >= NTOT) return;
    int c = (int)(idx & 127);
    size_t p = idx >> 7;
    int x = (int)(p % WWID);
    size_t t = p / WWID;
    int y = (int)(t % HH);
    int b = (int)(t / HH);

    float wHv[4], wWv[4];
    #pragma unroll
    for (int s = 0; s < 4; ++s) {
        wHv[s] = wH[(((size_t)b * 4 + s) * HH + y) * WWID + x];
        wWv[s] = wW[(((size_t)b * 4 + s) * HH + y) * WWID + x];
    }
    float acc = 0.f;
    #pragma unroll
    for (int shh = 0; shh < 4; ++shh) {
        int yy = y + shh; if (yy > HH - 1) yy = HH - 1;
        const float* row = af + ((size_t)(b * HH + yy) * WWID) * CC;
        #pragma unroll
        for (int sww = 0; sww < 4; ++sww) {
            int xx = x + sww; if (xx > WWID - 1) xx = WWID - 1;
            acc += wHv[shh] * wWv[sww] * row[(size_t)xx * CC + c];
        }
    }
    aligned[((size_t)(b * HH + y) * WWID + x) * CC + c] = acc;
}

// ---------------------------------------------------------------------------
// Point gather (NHWC -> contiguous per-point channels).
// ---------------------------------------------------------------------------
__global__ __launch_bounds__(256) void gather_kernel(
    const float* __restrict__ aligned, const int* __restrict__ pts,
    const int* __restrict__ stridePtr, float* __restrict__ out, int npts)
{
    int idx = blockIdx.x * 256 + threadIdx.x;
    if (idx >= npts * CC) return;
    int i = idx >> 7, c = idx & 127;
    int s = stridePtr[0];
    int b = pts[3 * i + 0];
    int h = pts[3 * i + 1] / s;
    int w = pts[3 * i + 2] / s;
    out[idx] = aligned[(((size_t)(b * HH + h)) * WWID + w) * CC + c];
}

// ---------------------------------------------------------------------------
extern "C" void kernel_launch(void* const* d_in, const int* in_sizes, int n_in,
                              void* d_out, int out_size, void* d_ws, size_t ws_size,
                              hipStream_t stream) {
    const float* x   = (const float*)d_in[0];
    const float* w1  = (const float*)d_in[1];
    const float* g1  = (const float*)d_in[2];
    const float* b1v = (const float*)d_in[3];
    const float* m1  = (const float*)d_in[4];
    const float* v1  = (const float*)d_in[5];
    const float* w2  = (const float*)d_in[6];
    const float* g2  = (const float*)d_in[7];
    const float* b2v = (const float*)d_in[8];
    const float* m2  = (const float*)d_in[9];
    const float* v2  = (const float*)d_in[10];
    const float* wh  = (const float*)d_in[11];
    const float* bh  = (const float*)d_in[12];
    const float* ww  = (const float*)d_in[13];
    const float* bw  = (const float*)d_in[14];
    const float* wf  = (const float*)d_in[15];
    const float* bfv = (const float*)d_in[16];
    const int* pts   = (const int*)d_in[17];
    const int* strd  = (const int*)d_in[18];
    float* out = (float*)d_out;
    const int npts = in_sizes[17] / 3;

    const size_t NXC = (size_t)BB * CC * HH * WWID;   // 11,534,336
    const size_t WTN = 9ull * CC * CC;                // 147,456

    char* ws = (char*)d_ws;
    size_t off = 0;
    auto take = [&](size_t bytes) -> char* {
        char* p = ws + off;
        off = (off + bytes + 255) & ~(size_t)255;
        return p;
    };
    unsigned short* XIN = (unsigned short*)take(NXC * 2);
    unsigned short* Y1  = (unsigned short*)take(NXC * 2);
    unsigned short* X2  = (unsigned short*)take(NXC * 2);
    float* AF  = (float*)take(NXC * 4);
    float* ALN = (float*)take(NXC * 4);
    float* WH  = (float*)take((size_t)BB * 4 * HH * WWID * 4);
    float* WW_ = (float*)take((size_t)BB * 4 * HH * WWID * 4);
    unsigned short* WT1 = (unsigned short*)take(WTN * 2);
    unsigned short* WT2 = (unsigned short*)take(WTN * 2);
    unsigned short* WFB = (unsigned short*)take((size_t)CC * CC * 2);
    float* SC1 = (float*)take(CC * 4);
    float* SH1 = (float*)take(CC * 4);
    float* SC2 = (float*)take(CC * 4);
    float* SH2 = (float*)take(CC * 4);

    float* lossPtr = out + (size_t)(out_size - 1);

    // 1) prep
    {
        unsigned long long total = NXC + 2 * WTN + (size_t)CC * CC + 2 * CC + 1;
        int blocks = (int)((total + 255) / 256);
        prep_kernel<<<blocks, 256, 0, stream>>>(
            x, w1, w2, g1, b1v, m1, v1, g2, b2v, m2, v2, wf,
            XIN, WT1, WT2, WFB, SC1, SH1, SC2, SH2, lossPtr, NXC);
    }

    dim3 tileGrid(WWID / 32, HH, BB);

    // 2) conv1 + bn + relu
    conv3_bn_relu_kernel<<<tileGrid, 256, 0, stream>>>(XIN, WT1, SC1, SH1, Y1);
    // 3) conv2 + bn + relu
    conv3_bn_relu_kernel<<<tileGrid, 256, 0, stream>>>(Y1, WT2, SC2, SH2, X2);
    // 4) head: af / softmax aggregates / loss
    head_kernel<<<tileGrid, 256, 0, stream>>>(X2, WFB, bfv, wh, bh, ww, bw,
                                              AF, WH, WW_, lossPtr);
    // 5) soft offset
    {
        int blocks = (int)((NXC + 255) / 256);
        soft_offset_kernel<<<blocks, 256, 0, stream>>>(AF, WH, WW_, ALN);
    }
    // 6) gather
    {
        int blocks = (npts * CC + 255) / 256;
        gather_kernel<<<blocks, 256, 0, stream>>>(ALN, pts, strd, out, npts);
    }
}
